// CrossEncoder_48498770707146
// MI455X (gfx1250) — matrix operand order repaired and checked by reading, not compiled, not run on hardware
//
#include <hip/hip_runtime.h>
#include <hip/hip_bf16.h>

// ---------------- problem constants ----------------
constexpr int Bd    = 2;
constexpr int Ed    = 512;
constexpr int Hd    = 8;
constexpr int DH    = 64;     // head dim
constexpr int MLPd  = 2048;
constexpr int NQd   = 2048;
constexpr int NKVd  = 2048;
constexpr int DEPTH = 2;

typedef __attribute__((ext_vector_type(16))) __bf16 v16bf;
typedef __attribute__((ext_vector_type(8)))  float  v8f;

union AF { v16bf v; unsigned int u[8]; };
union BF { v16bf v; unsigned int u[8]; };

__device__ __forceinline__ unsigned short f2bf(float f) {
    unsigned int u = __float_as_uint(f);
    unsigned int r = u + 0x7FFFu + ((u >> 16) & 1u);
    return (unsigned short)(r >> 16);
}

// ---------------- CDNA5 feature detection ----------------
#if defined(__has_builtin)
#  if __has_builtin(__builtin_amdgcn_global_load_async_to_lds_b128)
#    define HAVE_ASYNC 1
#  else
#    define HAVE_ASYNC 0
#  endif
#  if __has_builtin(__builtin_amdgcn_tensor_load_to_lds)
#    define HAVE_TDM 1
#  else
#    define HAVE_TDM 0
#  endif
#else
#  define HAVE_ASYNC 0
#  define HAVE_TDM 0
#endif

typedef int vi4 __attribute__((ext_vector_type(4)));
typedef __attribute__((address_space(1))) vi4 gvi4;   // global int4
typedef __attribute__((address_space(3))) vi4 lvi4;   // LDS int4

// 16B global -> LDS copy: async-to-LDS when available, else synchronous.
__device__ __forceinline__ void ld_tile_b128(const unsigned short* g, unsigned short* l) {
#if HAVE_ASYNC
    __builtin_amdgcn_global_load_async_to_lds_b128(
        (gvi4*)(unsigned long long)(const void*)g,
        (lvi4*)(unsigned int)(unsigned long long)(void*)l,
        0, 0);
#else
    *(uint4*)l = *(const uint4*)g;
#endif
}

__device__ __forceinline__ void wait_async() {
#if HAVE_ASYNC
#  if __has_builtin(__builtin_amdgcn_s_wait_asynccnt)
    __builtin_amdgcn_s_wait_asynccnt(0);
#  else
    asm volatile("s_wait_asynccnt 0x0" ::: "memory");
#  endif
#endif
}

#if HAVE_TDM
typedef unsigned int v4u_t __attribute__((ext_vector_type(4)));
typedef int          v8i_t __attribute__((ext_vector_type(8)));
typedef int          v4i_t __attribute__((ext_vector_type(4)));

// 2D bf16 tile load via Tensor Data Mover: tile_rows x tile_cols elements,
// global rows strided by row_stride elements, packed contiguously into LDS.
__device__ __forceinline__ void tdm_load_2d_bf16(const void* g, unsigned int lds_off,
                                                 int tile_cols, int tile_rows,
                                                 unsigned long long row_stride) {
    unsigned long long ga = (unsigned long long)g;
    v4u_t g0;
    g0.x = 1u;                                                    // count=1 (valid user D#)
    g0.y = lds_off;                                               // lds_addr (bytes)
    g0.z = (unsigned int)(ga & 0xFFFFFFFFu);                      // global_addr[31:0]
    g0.w = (unsigned int)((ga >> 32) & 0x01FFFFFFu) | (2u << 30); // addr[56:32] | type=2
    unsigned int td0 = (unsigned int)row_stride;                  // tensor_dim0
    unsigned int td1 = 1u << 20;                                  // tensor_dim1 (ample)
    v8i_t g1;
    g1[0] = (int)(1u << 16);                                      // data_size=1 -> 2 bytes
    g1[1] = (int)((td0 & 0xFFFFu) << 16);                         // tensor_dim0 lo
    g1[2] = (int)(((td0 >> 16) & 0xFFFFu) | ((td1 & 0xFFFFu) << 16)); // dim0 hi | dim1 lo
    g1[3] = (int)(((td1 >> 16) & 0xFFFFu) | (((unsigned)tile_cols & 0xFFFFu) << 16)); // dim1 hi | tile_dim0
    g1[4] = (int)((unsigned)tile_rows & 0xFFFFu);                 // tile_dim1 (tile_dim2=0)
    g1[5] = (int)(unsigned)(row_stride & 0xFFFFFFFFull);          // tensor_dim0_stride lo
    g1[6] = (int)(unsigned)((row_stride >> 32) & 0xFFFFull);      // stride hi | dim1_stride lo=0
    g1[7] = 0;
    v4i_t z4 = {0, 0, 0, 0};
#if __clang_major__ >= 23
    v8i_t z8 = {0, 0, 0, 0, 0, 0, 0, 0};
    __builtin_amdgcn_tensor_load_to_lds(g0, g1, z4, z4, z8, 0);
#else
    __builtin_amdgcn_tensor_load_to_lds(g0, g1, z4, z4, 0);
#endif
}
#endif // HAVE_TDM

// ---------------- fp32 -> bf16 convert ----------------
__global__ void cvt_f32_bf16(const float* __restrict__ in,
                             unsigned short* __restrict__ out, int n) {
    int i = blockIdx.x * blockDim.x + threadIdx.x;
    int stride = gridDim.x * blockDim.x;
    for (; i < n; i += stride) out[i] = f2bf(in[i]);
}

// ---------------- LayerNorm (row = E=512), out bf16 ----------------
__global__ __launch_bounds__(256) void ln_kernel(
    const float* __restrict__ x, unsigned short* __restrict__ o,
    const float* __restrict__ g, const float* __restrict__ bb) {
    int row = blockIdx.x;
    int tid = threadIdx.x;
    const float* xr = x + (size_t)row * Ed;
    float v0 = xr[tid], v1 = xr[tid + 256];
    float s = v0 + v1, s2 = v0 * v0 + v1 * v1;
#pragma unroll
    for (int off = 16; off > 0; off >>= 1) {
        s  += __shfl_xor(s,  off, 32);
        s2 += __shfl_xor(s2, off, 32);
    }
    __shared__ float w1s[8], w2s[8];
    int w = tid >> 5, lane = tid & 31;
    if (lane == 0) { w1s[w] = s; w2s[w] = s2; }
    __syncthreads();
    if (w == 0) {
        float a  = (lane < 8) ? w1s[lane] : 0.f;
        float b2 = (lane < 8) ? w2s[lane] : 0.f;
#pragma unroll
        for (int off = 4; off > 0; off >>= 1) {
            a  += __shfl_xor(a,  off, 32);
            b2 += __shfl_xor(b2, off, 32);
        }
        if (lane == 0) { w1s[0] = a; w2s[0] = b2; }
    }
    __syncthreads();
    float mu  = w1s[0] * (1.f / Ed);
    float var = w2s[0] * (1.f / Ed) - mu * mu;
    float ri  = rsqrtf(var + 1e-5f);
    o[(size_t)row * Ed + tid]       = f2bf((v0 - mu) * ri * g[tid]       + bb[tid]);
    o[(size_t)row * Ed + tid + 256] = f2bf((v1 - mu) * ri * g[tid + 256] + bb[tid + 256]);
}

// ---------------- bf16 WMMA GEMM: C = A(MxK) * B(KxN) ----------------
// EPI: 0 = store bf16 ; 1 = +bias, GELU, store bf16 ; 2 = +bias, += fp32 residual
constexpr int BM = 128, BN = 128, BK = 32;

template <int EPI>
__global__ __launch_bounds__(256) void gemm_bf16(
    const unsigned short* __restrict__ A, int lda,
    const unsigned short* __restrict__ B, int ldb,
    const float* __restrict__ bias,
    unsigned short* __restrict__ outb, int ldo,
    float* __restrict__ resid, int ldr, int K) {
    __shared__ unsigned short As[2][BM][BK];
    __shared__ unsigned short Bs[2][BK][BN];
    int tid  = threadIdx.x;
    int lane = tid & 31, w = tid >> 5;
    int wm = w & 3, wn = w >> 2;              // 4 waves along M, 2 along N
    int bm = blockIdx.y * BM, bn = blockIdx.x * BN;
    int Mh = lane & 15, half = lane >> 4;

    v8f acc[2][4];
#pragma unroll
    for (int mi = 0; mi < 2; mi++)
#pragma unroll
        for (int ni = 0; ni < 4; ni++) acc[mi][ni] = (v8f){0,0,0,0,0,0,0,0};

    auto stage = [&](int kb, int bsel) {
#pragma unroll
        for (int i = tid; i < (BM * BK / 8); i += 256) {      // A tile: 128x32
            int row = i >> 2, seg = i & 3;
            ld_tile_b128(A + (size_t)(bm + row) * lda + kb + seg * 8,
                         &As[bsel][row][seg * 8]);
        }
#pragma unroll
        for (int i = tid; i < (BK * BN / 8); i += 256) {      // B tile: 32x128
            int row = i >> 4, seg = i & 15;
            ld_tile_b128(B + (size_t)(kb + row) * ldb + bn + seg * 8,
                         &Bs[bsel][row][seg * 8]);
        }
    };

    const int nk = K / BK;
    stage(0, 0);
    wait_async();
    __syncthreads();
    int buf = 0;

    for (int t = 0; t < nk; t++) {
        if (t + 1 < nk) stage((t + 1) * BK, buf ^ 1);   // prefetch next tile (other buffer)

        AF a[2];
#pragma unroll
        for (int mi = 0; mi < 2; mi++) {
            int m0 = wm * 32 + mi * 16 + Mh;
#pragma unroll
            for (int j = 0; j < 8; j++) {
                int kk = (j < 4) ? (j * 2 + half * 8) : ((j - 4) * 2 + half * 8 + 16);
                a[mi].u[j] = *(const unsigned int*)&As[buf][m0][kk];
            }
        }
        BF bfv[4];
#pragma unroll
        for (int ni = 0; ni < 4; ni++) {
            int n0 = wn * 64 + ni * 16 + Mh;
#pragma unroll
            for (int j = 0; j < 8; j++) {
                int k0 = (j < 4) ? (j * 2 + half * 8) : ((j - 4) * 2 + half * 8 + 16);
                unsigned int lo = Bs[buf][k0][n0], hi = Bs[buf][k0 + 1][n0];
                bfv[ni].u[j] = lo | (hi << 16);
            }
        }
#pragma unroll
        for (int mi = 0; mi < 2; mi++)
#pragma unroll
            for (int ni = 0; ni < 4; ni++)
                acc[mi][ni] = __builtin_amdgcn_wmma_f32_16x16x32_bf16(
                    false, a[mi].v, false, bfv[ni].v, (short)0, acc[mi][ni], false, false);

        wait_async();          // prefetched tile resident in LDS
        __syncthreads();       // everyone done reading buf; flip
        buf ^= 1;
    }

    // epilogue
#pragma unroll
    for (int mi = 0; mi < 2; mi++)
#pragma unroll
        for (int ni = 0; ni < 4; ni++) {
            int colg = bn + wn * 64 + ni * 16 + Mh;
            float bv = (EPI != 0) ? bias[colg] : 0.f;
#pragma unroll
            for (int r = 0; r < 8; r++) {
                int rowg = bm + wm * 32 + mi * 16 + r + half * 8;
                float v = acc[mi][ni][r];
                if (EPI == 0) {
                    outb[(size_t)rowg * ldo + colg] = f2bf(v);
                } else if (EPI == 1) {
                    float x = v + bv;
                    float gl = 0.5f * x * (1.f + erff(x * 0.70710678118f));
                    outb[(size_t)rowg * ldo + colg] = f2bf(gl);
                } else {
                    float* p = resid + (size_t)rowg * ldr + colg;
                    *p = *p + v + bv;
                }
            }
        }
}

// ---------------- flash attention (per b,h ; 128 queries / block) ----------------
__global__ __launch_bounds__(256) void attn_kernel(
    const unsigned short* __restrict__ Q, const unsigned short* __restrict__ Kb,
    const unsigned short* __restrict__ Vb, unsigned short* __restrict__ O) {
    __shared__ unsigned short Ks[32][64];
    __shared__ unsigned short Vs[32][64];
    __shared__ unsigned short Ps[8][16][32];   // per-wave P scratch (C-layout -> A-layout)

    int tid = threadIdx.x, lane = tid & 31, w = tid >> 5;
    int bh = blockIdx.y;
    int b = bh >> 3, h = bh & 7;
    int q0 = blockIdx.x * 128 + w * 16;
    int Mh = lane & 15, half = lane >> 4;
    const float scale = 0.125f;   // 1/sqrt(64)

    // load Q fragments (2 chunks of K=32 over d=64), stays in registers
    AF qa[2];
    const unsigned short* qrow = Q + ((size_t)(b * NQd + q0 + Mh)) * Ed + h * DH;
#pragma unroll
    for (int c = 0; c < 2; c++)
#pragma unroll
        for (int j = 0; j < 8; j++) {
            int kk = c * 32 + ((j < 4) ? (j * 2 + half * 8) : ((j - 4) * 2 + half * 8 + 16));
            qa[c].u[j] = *(const unsigned int*)(qrow + kk);
        }

    v8f acc[4];
#pragma unroll
    for (int t = 0; t < 4; t++) acc[t] = (v8f){0,0,0,0,0,0,0,0};
    float mrow[8], lrow[8];
#pragma unroll
    for (int r = 0; r < 8; r++) { mrow[r] = -1e30f; lrow[r] = 0.f; }

#if HAVE_TDM
    unsigned int ldsK = (unsigned int)(unsigned long long)(void*)&Ks[0][0];
    unsigned int ldsV = (unsigned int)(unsigned long long)(void*)&Vs[0][0];
#endif

    for (int kv0 = 0; kv0 < NKVd; kv0 += 32) {
        __syncthreads();   // prior iteration finished reading Ks/Vs/Ps
#if HAVE_TDM
        if (w == 0) {      // one TDM descriptor pair per workgroup (wave-level op)
            const unsigned short* kg = Kb + ((size_t)(b * NKVd + kv0)) * Ed + h * DH;
            const unsigned short* vg = Vb + ((size_t)(b * NKVd + kv0)) * Ed + h * DH;
            tdm_load_2d_bf16(kg, ldsK, 64, 32, (unsigned long long)Ed);
            tdm_load_2d_bf16(vg, ldsV, 64, 32, (unsigned long long)Ed);
            __builtin_amdgcn_s_wait_tensorcnt(0);
        }
        __syncthreads();
#else
        {   // cooperative load of 32 keys/values (32x64 bf16 each): one 16B copy per thread
            int row = tid >> 3, seg = tid & 7;
            size_t src = ((size_t)(b * NKVd + kv0 + row)) * Ed + h * DH + seg * 8;
            ld_tile_b128(Kb + src, &Ks[row][seg * 8]);
            ld_tile_b128(Vb + src, &Vs[row][seg * 8]);
        }
        wait_async();
        __syncthreads();
#endif

        // scores: two 16-key tiles, each accumulated over two d-chunks
        v8f s[2];
        s[0] = (v8f){0,0,0,0,0,0,0,0};
        s[1] = (v8f){0,0,0,0,0,0,0,0};
#pragma unroll
        for (int kt = 0; kt < 2; kt++) {
            BF kf[2];
#pragma unroll
            for (int c = 0; c < 2; c++)
#pragma unroll
                for (int j = 0; j < 8; j++) {
                    int k0 = (j < 4) ? (j * 2 + half * 8) : ((j - 4) * 2 + half * 8 + 16);                 // d index in chunk
                    kf[c].u[j] = *(const unsigned int*)&Ks[kt * 16 + Mh][c * 32 + k0];
                }
            s[kt] = __builtin_amdgcn_wmma_f32_16x16x32_bf16(
                false, qa[0].v, false, kf[0].v, (short)0, s[kt], false, false);
            s[kt] = __builtin_amdgcn_wmma_f32_16x16x32_bf16(
                false, qa[1].v, false, kf[1].v, (short)0, s[kt], false, false);
        }

        // online softmax update
        float fac[8];
#pragma unroll
        for (int r = 0; r < 8; r++) {
            float a0 = s[0][r] * scale, a1 = s[1][r] * scale;
            float mx = fmaxf(a0, a1);
#pragma unroll
            for (int off = 1; off < 16; off <<= 1) mx = fmaxf(mx, __shfl_xor(mx, off, 16));
            float mnew = fmaxf(mrow[r], mx);
            fac[r] = __expf(mrow[r] - mnew);
            mrow[r] = mnew;
            float p0 = __expf(a0 - mnew), p1 = __expf(a1 - mnew);
            s[0][r] = p0; s[1][r] = p1;
            float sv = p0 + p1;
#pragma unroll
            for (int off = 1; off < 16; off <<= 1) sv += __shfl_xor(sv, off, 16);
            lrow[r] = lrow[r] * fac[r] + sv;
        }
#pragma unroll
        for (int t = 0; t < 4; t++)
#pragma unroll
            for (int r = 0; r < 8; r++) acc[t][r] *= fac[r];

        // spill P (C-layout) to LDS as bf16
#pragma unroll
        for (int kt = 0; kt < 2; kt++)
#pragma unroll
            for (int r = 0; r < 8; r++)
                Ps[w][r + half * 8][kt * 16 + Mh] = f2bf(s[kt][r]);
        __syncthreads();

        // reload P in A-layout, multiply by V
        AF pf;
#pragma unroll
        for (int j = 0; j < 8; j++) {
            int kk = (j < 4) ? (j * 2 + half * 8) : ((j - 4) * 2 + half * 8 + 16);
            pf.u[j] = *(const unsigned int*)&Ps[w][Mh][kk];
        }
#pragma unroll
        for (int t = 0; t < 4; t++) {
            BF vf;
#pragma unroll
            for (int j = 0; j < 8; j++) {
                int k0 = (j < 4) ? (j * 2 + half * 8) : ((j - 4) * 2 + half * 8 + 16);                     // key index
                unsigned int lo = Vs[k0][t * 16 + Mh];
                unsigned int hi = Vs[k0 + 1][t * 16 + Mh];
                vf.u[j] = lo | (hi << 16);
            }
            acc[t] = __builtin_amdgcn_wmma_f32_16x16x32_bf16(
                false, pf.v, false, vf.v, (short)0, acc[t], false, false);
        }
    }

    // write normalized output
#pragma unroll
    for (int t = 0; t < 4; t++)
#pragma unroll
        for (int r = 0; r < 8; r++) {
            int row = q0 + r + half * 8;
            int col = h * DH + t * 16 + Mh;
            O[((size_t)(b * NQd + row)) * Ed + col] = f2bf(acc[t][r] / lrow[r]);
        }
}

// ---------------- host orchestration ----------------
extern "C" void kernel_launch(void* const* d_in, const int* in_sizes, int n_in,
                              void* d_out, int out_size, void* d_ws, size_t ws_size,
                              hipStream_t stream) {
    const float* x_q   = (const float*)d_in[0];
    const float* x_kv  = (const float*)d_in[1];
    const float* wq    = (const float*)d_in[2];
    const float* wkv   = (const float*)d_in[3];
    const float* wo    = (const float*)d_in[4];
    const float* bo    = (const float*)d_in[5];
    const float* w1    = (const float*)d_in[6];
    const float* b1    = (const float*)d_in[7];
    const float* w2    = (const float*)d_in[8];
    const float* b2    = (const float*)d_in[9];
    const float* ln1_g = (const float*)d_in[10];
    const float* ln1_b = (const float*)d_in[11];
    const float* ln2_g = (const float*)d_in[12];
    const float* ln2_b = (const float*)d_in[13];

    const size_t ACT = (size_t)Bd * NQd * Ed;          // 2,097,152
    char* ws = (char*)d_ws;
    size_t off = 0;
    float*          xq   = (float*)(ws + off);          off += ACT * 4;          // fp32 running x_q
    unsigned short* hq   = (unsigned short*)(ws + off); off += ACT * 2;
    unsigned short* hkv  = (unsigned short*)(ws + off); off += ACT * 2;
    unsigned short* qb   = (unsigned short*)(ws + off); off += ACT * 2;
    unsigned short* kb   = (unsigned short*)(ws + off); off += ACT * 2;
    unsigned short* vb   = (unsigned short*)(ws + off); off += ACT * 2;
    unsigned short* ao   = (unsigned short*)(ws + off); off += ACT * 2;
    unsigned short* ff1  = (unsigned short*)(ws + off); off += (size_t)Bd * NQd * MLPd * 2;
    unsigned short* wq_bf  = (unsigned short*)(ws + off); off += (size_t)DEPTH * Ed * Ed * 2;
    unsigned short* wkv_bf = (unsigned short*)(ws + off); off += (size_t)DEPTH * Ed * 2 * Ed * 2;
    unsigned short* wo_bf  = (unsigned short*)(ws + off); off += (size_t)DEPTH * Ed * Ed * 2;
    unsigned short* w1_bf  = (unsigned short*)(ws + off); off += (size_t)DEPTH * Ed * MLPd * 2;
    unsigned short* w2_bf  = (unsigned short*)(ws + off); off += (size_t)DEPTH * MLPd * Ed * 2;

    // weights fp32 -> bf16
    cvt_f32_bf16<<<1024, 256, 0, stream>>>(wq,  wq_bf,  DEPTH * Ed * Ed);
    cvt_f32_bf16<<<1024, 256, 0, stream>>>(wkv, wkv_bf, DEPTH * Ed * 2 * Ed);
    cvt_f32_bf16<<<1024, 256, 0, stream>>>(wo,  wo_bf,  DEPTH * Ed * Ed);
    cvt_f32_bf16<<<1024, 256, 0, stream>>>(w1,  w1_bf,  DEPTH * Ed * MLPd);
    cvt_f32_bf16<<<1024, 256, 0, stream>>>(w2,  w2_bf,  DEPTH * MLPd * Ed);

    (void)hipMemcpyAsync(xq, x_q, ACT * sizeof(float), hipMemcpyDeviceToDevice, stream);

    const int ROWS = Bd * NQd;                         // 4096
    dim3 gN512(Ed / BN,  ROWS / BM);                   // (4, 32)
    dim3 gN2048(MLPd / BN, ROWS / BM);                 // (16, 32)
    dim3 gAttn(NQd / 128, Bd * Hd);                    // (16, 16)

    for (int l = 0; l < DEPTH; l++) {
        const unsigned short* wql  = wq_bf  + (size_t)l * Ed * Ed;
        const unsigned short* wkvl = wkv_bf + (size_t)l * Ed * 2 * Ed;
        const unsigned short* wol  = wo_bf  + (size_t)l * Ed * Ed;
        const unsigned short* w1l  = w1_bf  + (size_t)l * Ed * MLPd;
        const unsigned short* w2l  = w2_bf  + (size_t)l * MLPd * Ed;

        // prenorm (same LN params for q and kv streams)
        ln_kernel<<<ROWS, 256, 0, stream>>>(xq,   hq,  ln1_g + l * Ed, ln1_b + l * Ed);
        ln_kernel<<<ROWS, 256, 0, stream>>>(x_kv, hkv, ln1_g + l * Ed, ln1_b + l * Ed);

        // projections
        gemm_bf16<0><<<gN512, 256, 0, stream>>>(hq,  Ed, wql,       Ed,       nullptr, qb, Ed, nullptr, 0, Ed);
        gemm_bf16<0><<<gN512, 256, 0, stream>>>(hkv, Ed, wkvl,      2 * Ed,   nullptr, kb, Ed, nullptr, 0, Ed);
        gemm_bf16<0><<<gN512, 256, 0, stream>>>(hkv, Ed, wkvl + Ed, 2 * Ed,   nullptr, vb, Ed, nullptr, 0, Ed);

        // attention
        attn_kernel<<<gAttn, 256, 0, stream>>>(qb, kb, vb, ao);

        // output projection + bias + residual into xq
        gemm_bf16<2><<<gN512, 256, 0, stream>>>(ao, Ed, wol, Ed, bo + l * Ed, nullptr, 0, xq, Ed, Ed);

        // FFN
        ln_kernel<<<ROWS, 256, 0, stream>>>(xq, hq, ln2_g + l * Ed, ln2_b + l * Ed);
        gemm_bf16<1><<<gN2048, 256, 0, stream>>>(hq, Ed, w1l, MLPd, b1 + l * MLPd, ff1, MLPd, nullptr, 0, Ed);
        gemm_bf16<2><<<gN512,  256, 0, stream>>>(ff1, MLPd, w2l, Ed, b2 + l * Ed, nullptr, 0, xq, Ed, MLPd);
    }

    (void)hipMemcpyAsync(d_out, xq, ACT * sizeof(float), hipMemcpyDeviceToDevice, stream);
}
